// Model_63556926046619
// MI455X (gfx1250) — matrix operand order repaired and checked by reading, not compiled, not run on hardware
//
#include <hip/hip_runtime.h>
#include <hip/hip_bf16.h>
#include <math.h>
#include <stdint.h>

// ---------------- types for CDNA5 WMMA ----------------
typedef __bf16 bf16_t;
typedef __attribute__((ext_vector_type(16))) bf16_t v16bf;
typedef __attribute__((ext_vector_type(8)))  float  v8f;

#define IN_F   13824
#define HID    1024
#define LH     512
#define G4     2048
#define NCLS   4787
#define SEQ    512
#define LMAX   32
#define BN_EPS 1e-5f

__device__ __forceinline__ unsigned short f2bf(float f) {
  __bf16 b = (__bf16)f;                       // RNE; lowers to v_cvt_*bf16_f32
  return __builtin_bit_cast(unsigned short, b);
}
__device__ __forceinline__ unsigned pk2bf(float lo, float hi) {
  return (unsigned)f2bf(lo) | ((unsigned)f2bf(hi) << 16);
}

__device__ __forceinline__ v8f v8f_zero() {
  v8f z;
#pragma unroll
  for (int i = 0; i < 8; ++i) z[i] = 0.f;
  return z;
}

// ---- CDNA5 async global->LDS copy (ASYNCcnt), ISA §10.7 / §15.18.3 ----
__device__ __forceinline__ void async_ld_b128(void* lds_ptr, const void* gptr) {
  unsigned lds_off = (unsigned)(uintptr_t)lds_ptr;   // low 32 bits = LDS offset
  unsigned long long ga = (unsigned long long)(uintptr_t)gptr;
  asm volatile("global_load_async_to_lds_b128 %0, %1, off"
               :: "v"(lds_off), "v"(ga) : "memory");
}
__device__ __forceinline__ void wait_async0() {
  asm volatile("s_wait_asynccnt 0" ::: "memory");
}

union FragAB { v16bf v; uint4 q[2]; unsigned short s[16]; };

// =====================================================================
// Generic GEMM: C[m][n] = sum_k A[m][k]*W[n][k] (+bias0[n] (+bias1[n]))
// A: f32 (inline-converted) or bf16 (async global->LDS); W: f32 NxK.
// Block tile 128x128, BK=64, double-buffered LDS: stage k+1 is issued
// before computing stage k; ASYNCcnt drained only after the WMMA chain.
// OOB tile rows/cols are CLAMPED (not zeroed): garbage rows/cols only
// influence C rows/cols that the guarded epilogue never stores.
// =====================================================================
template<bool A_BF16, bool RELU, bool BIAS1>
__global__ __launch_bounds__(256)
void gemm_wmma_bf16(const void* __restrict__ Av, int lda,
                    const float* __restrict__ W, int ldw,
                    float* __restrict__ C, int ldc,
                    int M, int N, int K,
                    const float* __restrict__ bias0,
                    const float* __restrict__ bias1)
{
  __shared__ __align__(16) unsigned short As[2][128][72];
  __shared__ __align__(16) unsigned short Bs[2][128][72];
  const int tid  = threadIdx.x;
  const int lane = tid & 31;
  const int wv   = tid >> 5;       // 0..7
  const int waveRow = wv >> 1;     // 0..3 -> 32-row strips
  const int waveCol = wv & 1;      // 0..1 -> 64-col strips
  const int m0 = blockIdx.y * 128;
  const int n0 = blockIdx.x * 128;

  v8f acc[2][4];
#pragma unroll
  for (int i = 0; i < 2; ++i)
#pragma unroll
    for (int j = 0; j < 4; ++j) acc[i][j] = v8f_zero();

  auto stage = [&](int buf, int k0) {
    if (A_BF16) {
      const unsigned short* A = (const unsigned short*)Av;
#pragma unroll
      for (int i = 0; i < 4; ++i) {
        int lin = tid + i * 256;            // 1024 uint4
        int r = lin >> 3, c8 = (lin & 7) * 8;
        int m = m0 + r; if (m >= M) m = M - 1;     // clamp: discarded later
        async_ld_b128(&As[buf][r][c8], A + (size_t)m * lda + k0 + c8);
      }
    } else {
      const float* A = (const float*)Av;
#pragma unroll
      for (int i = 0; i < 8; ++i) {
        int lin = tid + i * 256;            // 2048 float4
        int r = lin >> 4, c4 = (lin & 15) * 4;
        int m = m0 + r; if (m >= M) m = M - 1;
        float4 v = *(const float4*)(A + (size_t)m * lda + k0 + c4);
        uint2 pk; pk.x = pk2bf(v.x, v.y); pk.y = pk2bf(v.z, v.w);
        *(uint2*)&As[buf][r][c4] = pk;
      }
    }
#pragma unroll
    for (int i = 0; i < 8; ++i) {           // B from W (NxK f32), inline cvt
      int lin = tid + i * 256;
      int r = lin >> 4, c4 = (lin & 15) * 4;
      int nn = n0 + r; if (nn >= N) nn = N - 1;
      float4 v = *(const float4*)(W + (size_t)nn * ldw + k0 + c4);
      uint2 pk; pk.x = pk2bf(v.x, v.y); pk.y = pk2bf(v.z, v.w);
      *(uint2*)&Bs[buf][r][c4] = pk;
    }
  };

  const int KT = K / 64;
  stage(0, 0);
  if (A_BF16) wait_async0();
  __syncthreads();

  const int half = lane >> 4;
  const int l16  = lane & 15;

  for (int kt = 0; kt < KT; ++kt) {
    const int buf = kt & 1;
    if (kt + 1 < KT) stage(buf ^ 1, (kt + 1) * 64);

#pragma unroll
    for (int ks = 0; ks < 2; ++ks) {
      FragAB af[2]; FragAB bfr[4];
#pragma unroll
      for (int rt = 0; rt < 2; ++rt) {
        int row = waveRow * 32 + rt * 16 + l16;
        int kb = ks * 32 + half * 8;      // A layout: elems0-7 K=half*8.., 8-15 K=16+half*8..
        af[rt].q[0] = *(const uint4*)&As[buf][row][kb];
        af[rt].q[1] = *(const uint4*)&As[buf][row][kb + 16];
      }
#pragma unroll
      for (int ct = 0; ct < 4; ++ct) {
        int row = waveCol * 64 + ct * 16 + l16;
        int kb = ks * 32 + half * 8;      // B layout: elems0-7 K=half*8.., 8-15 K=16+half*8..
        bfr[ct].q[0] = *(const uint4*)&Bs[buf][row][kb];
        bfr[ct].q[1] = *(const uint4*)&Bs[buf][row][kb + 16];
      }
#pragma unroll
      for (int rt = 0; rt < 2; ++rt)
#pragma unroll
        for (int ct = 0; ct < 4; ++ct)
          acc[rt][ct] = __builtin_amdgcn_wmma_f32_16x16x32_bf16(
              false, af[rt].v, false, bfr[ct].v, (short)0, acc[rt][ct], false, false);
    }
    if (A_BF16) wait_async0();   // drain AFTER compute: latency hidden
    __syncthreads();
  }

  // ---- epilogue: bias (+bias1), optional relu, guarded store ----
  const int rowOff = (lane >> 4) * 8;
#pragma unroll
  for (int ct = 0; ct < 4; ++ct) {
    int col = n0 + waveCol * 64 + ct * 16 + l16;
    if (col >= N) continue;
    float b = bias0 ? bias0[col] : 0.f;
    if (BIAS1) b += bias1[col];
#pragma unroll
    for (int rt = 0; rt < 2; ++rt) {
#pragma unroll
      for (int e = 0; e < 8; ++e) {
        int row = m0 + waveRow * 32 + rt * 16 + rowOff + e;
        if (row < M) {
          float v = acc[rt][ct][e] + b;
          if (RELU) v = fmaxf(v, 0.f);
          C[(size_t)row * ldc + col] = v;
        }
      }
    }
  }
}

// =====================================================================
// LSTM step: per block, 64 seqs x 64 j-cols, all 4 gate chunks.
// g = h_prev @ w_hh^T (WMMA bf16) + xg[:,t,:]; fused cell update.
// Both operand tiles are bf16 -> double-buffered async global->LDS.
// =====================================================================
__global__ __launch_bounds__(256)
void lstm_step(const unsigned short* __restrict__ hin_base,
               unsigned short* __restrict__ hout_base,
               float* __restrict__ c_base,
               const float* __restrict__ xg_base,
               const unsigned short* __restrict__ whh_base,
               float* __restrict__ hs_base,
               int t)
{
  __shared__ __align__(16) unsigned short As[2][64][72];
  __shared__ __align__(16) unsigned short Bs[2][256][72];
  const int d = blockIdx.z;
  const unsigned short* hin = hin_base + (size_t)d * LH * LH;
  unsigned short* hout      = hout_base + (size_t)d * LH * LH;
  float* cst                = c_base + (size_t)d * LH * LH;
  const float* xg           = xg_base + (size_t)d * SEQ * LMAX * G4;
  const unsigned short* whh = whh_base + (size_t)d * G4 * LH;
  float* hs                 = hs_base + (size_t)d * SEQ * LMAX * LH;

  const int tid  = threadIdx.x;
  const int lane = tid & 31;
  const int wv   = tid >> 5;
  const int waveRow = wv >> 2;    // 0..1 (32-row strips)
  const int waveJ   = wv & 3;     // 0..3 (16-col strips)
  const int s0 = blockIdx.y * 64;
  const int j0 = blockIdx.x * 64;

  v8f acc[2][4];
#pragma unroll
  for (int i = 0; i < 2; ++i)
#pragma unroll
    for (int q = 0; q < 4; ++q) acc[i][q] = v8f_zero();

  auto stage = [&](int buf, int k0) {
#pragma unroll
    for (int i = 0; i < 2; ++i) {      // 64x64 bf16 = 512 uint4
      int lin = tid + i * 256;
      int r = lin >> 3, c8 = (lin & 7) * 8;
      async_ld_b128(&As[buf][r][c8], hin + (size_t)(s0 + r) * LH + k0 + c8);
    }
#pragma unroll
    for (int i = 0; i < 8; ++i) {      // 256x64 bf16 = 2048 uint4
      int lin = tid + i * 256;
      int r = lin >> 3, c8 = (lin & 7) * 8;
      int q = r >> 6, jl = r & 63;
      async_ld_b128(&Bs[buf][r][c8], whh + (size_t)(q * LH + j0 + jl) * LH + k0 + c8);
    }
  };

  const int KT = LH / 64;
  stage(0, 0);
  wait_async0();
  __syncthreads();

  const int half = lane >> 4, l16 = lane & 15;
  for (int kt = 0; kt < KT; ++kt) {
    const int buf = kt & 1;
    if (kt + 1 < KT) stage(buf ^ 1, (kt + 1) * 64);

#pragma unroll
    for (int ks = 0; ks < 2; ++ks) {
      FragAB af[2], bfr[4];
#pragma unroll
      for (int rt = 0; rt < 2; ++rt) {
        int row = waveRow * 32 + rt * 16 + l16;
        int kb = ks * 32 + half * 8;
        af[rt].q[0] = *(const uint4*)&As[buf][row][kb];
        af[rt].q[1] = *(const uint4*)&As[buf][row][kb + 16];
      }
#pragma unroll
      for (int q = 0; q < 4; ++q) {
        int row = q * 64 + waveJ * 16 + l16;
        int kb = ks * 32 + half * 8;
        bfr[q].q[0] = *(const uint4*)&Bs[buf][row][kb];
        bfr[q].q[1] = *(const uint4*)&Bs[buf][row][kb + 16];
      }
#pragma unroll
      for (int rt = 0; rt < 2; ++rt)
#pragma unroll
        for (int q = 0; q < 4; ++q)
          acc[rt][q] = __builtin_amdgcn_wmma_f32_16x16x32_bf16(
              false, af[rt].v, false, bfr[q].v, (short)0, acc[rt][q], false, false);
    }
    wait_async0();               // drain AFTER compute: latency hidden
    __syncthreads();
  }

  // fused cell update: i,f,g,o for the same (s,j) sit in matching slots
  const int rowOff = (lane >> 4) * 8;
  const int j = j0 + waveJ * 16 + l16;
#pragma unroll
  for (int rt = 0; rt < 2; ++rt) {
#pragma unroll
    for (int e = 0; e < 8; ++e) {
      int s = s0 + waveRow * 32 + rt * 16 + rowOff + e;
      size_t gb = (size_t)(s * LMAX + t) * G4 + j;
      float gi = acc[rt][0][e] + xg[gb];
      float gf = acc[rt][1][e] + xg[gb + LH];
      float gg = acc[rt][2][e] + xg[gb + 2 * LH];
      float go = acc[rt][3][e] + xg[gb + 3 * LH];
      float cp = cst[(size_t)s * LH + j];
      float si = 1.f / (1.f + __expf(-gi));
      float sf = 1.f / (1.f + __expf(-gf));
      float so = 1.f / (1.f + __expf(-go));
      float cn = sf * cp + si * tanhf(gg);
      float hn = so * tanhf(cn);
      cst[(size_t)s * LH + j] = cn;
      hs[(size_t)(s * LMAX + t) * LH + j] = hn;
      hout[(size_t)s * LH + j] = f2bf(hn);
    }
  }
}

// ---------------- BN stats (coalesced partials + atomics) ----------------
__global__ __launch_bounds__(256)
void col_stats(const float* __restrict__ h, int M,
               float* __restrict__ colsum, float* __restrict__ colsq)
{
  const int c  = threadIdx.x * 4;
  const int r0 = blockIdx.x * 128;
  const int r1 = (r0 + 128 < M) ? (r0 + 128) : M;
  float s0 = 0, s1 = 0, s2 = 0, s3 = 0, q0 = 0, q1 = 0, q2 = 0, q3 = 0;
  for (int r = r0; r < r1; ++r) {
    float4 v = *(const float4*)(h + (size_t)r * HID + c);
    s0 += v.x; s1 += v.y; s2 += v.z; s3 += v.w;
    q0 += v.x * v.x; q1 += v.y * v.y; q2 += v.z * v.z; q3 += v.w * v.w;
  }
  atomicAdd(colsum + c + 0, s0); atomicAdd(colsum + c + 1, s1);
  atomicAdd(colsum + c + 2, s2); atomicAdd(colsum + c + 3, s3);
  atomicAdd(colsq + c + 0, q0); atomicAdd(colsq + c + 1, q1);
  atomicAdd(colsq + c + 2, q2); atomicAdd(colsq + c + 3, q3);
}

__global__ void bn_finalize(const float* __restrict__ colsum,
                            const float* __restrict__ colsq,
                            const float* __restrict__ gamma,
                            const float* __restrict__ beta,
                            float* __restrict__ scale,
                            float* __restrict__ shift, float invM)
{
  int c = blockIdx.x * 256 + threadIdx.x;
  float mean = colsum[c] * invM;
  float var  = colsq[c] * invM - mean * mean;
  float sc   = gamma[c] * rsqrtf(var + BN_EPS);
  scale[c] = sc;
  shift[c] = beta[c] - mean * sc;
}

// ---------------- gather + normalize into bf16 seq_x / seq_rev ----------------
__global__ __launch_bounds__(256)
void gather_norm(const float* __restrict__ h, const float* __restrict__ scale,
                 const float* __restrict__ shift, const int* __restrict__ seq_idx,
                 const int* __restrict__ lengths,
                 unsigned short* __restrict__ seqx,
                 unsigned short* __restrict__ seqrev)
{
  const int t = blockIdx.x, s = blockIdx.y;
  const int len = lengths[s];
  const int g1 = (t < len) ? seq_idx[s * LMAX + t] : -1;
  int rt = len - 1 - t; if (rt < 0) rt = 0;
  const int g2 = seq_idx[s * LMAX + rt];
  const int c = threadIdx.x * 4;
  const size_t ob = (size_t)(s * LMAX + t) * HID + c;
  float4 sc = *(const float4*)(scale + c);
  float4 sh = *(const float4*)(shift + c);
  uint2 z; z.x = 0; z.y = 0;
  if (g1 >= 0) {
    float4 v = *(const float4*)(h + (size_t)g1 * HID + c);
    uint2 pk; pk.x = pk2bf(v.x * sc.x + sh.x, v.y * sc.y + sh.y);
    pk.y = pk2bf(v.z * sc.z + sh.z, v.w * sc.w + sh.w);
    *(uint2*)(seqx + ob) = pk;
  } else {
    *(uint2*)(seqx + ob) = z;
  }
  float4 v2 = *(const float4*)(h + (size_t)g2 * HID + c);
  uint2 pk2; pk2.x = pk2bf(v2.x * sc.x + sh.x, v2.y * sc.y + sh.y);
  pk2.y = pk2bf(v2.z * sc.z + sh.z, v2.w * sc.w + sh.w);
  *(uint2*)(seqrev + ob) = pk2;
}

// ---------------- scatter LSTM outputs into bf16 lstm_feats ----------------
__global__ __launch_bounds__(256)
void scatter_out(const float* __restrict__ hsf, const float* __restrict__ hsb,
                 const int* __restrict__ seq_idx, const int* __restrict__ lengths,
                 unsigned short* __restrict__ lf)
{
  const int t = blockIdx.x, s = blockIdx.y;
  const int len = lengths[s];
  if (t >= len) return;
  const int row = seq_idx[s * LMAX + t];
  const int rt = len - 1 - t;
  const int c = threadIdx.x * 4;
  float4 v;
  if (c < LH) v = *(const float4*)(hsf + (size_t)(s * LMAX + t) * LH + c);
  else        v = *(const float4*)(hsb + (size_t)(s * LMAX + rt) * LH + (c - LH));
  uint2 pk; pk.x = pk2bf(v.x, v.y); pk.y = pk2bf(v.z, v.w);
  *(uint2*)(lf + (size_t)row * HID + c) = pk;
}

// ---------------- prep utilities ----------------
__global__ void zero_f32k(float* p, int n) {
  int i = blockIdx.x * 256 + threadIdx.x;
  if (i < n) p[i] = 0.f;
}
__global__ void zero_u16(unsigned short* p, int n) {
  int i = blockIdx.x * 256 + threadIdx.x;
  if (i < n) p[i] = 0;
}
__global__ void cvt_bf16(const float* __restrict__ s, unsigned short* __restrict__ d, int n) {
  int i = blockIdx.x * 256 + threadIdx.x;
  if (i < n) d[i] = f2bf(s[i]);
}

// =====================================================================
extern "C" void kernel_launch(void* const* d_in, const int* in_sizes, int n_in,
                              void* d_out, int out_size, void* d_ws, size_t ws_size,
                              hipStream_t stream)
{
  const float* x      = (const float*)d_in[0];
  const float* fc1_w  = (const float*)d_in[1];
  const float* fc1_b  = (const float*)d_in[2];
  const float* gamma  = (const float*)d_in[3];
  const float* beta   = (const float*)d_in[4];
  const float* fc2_w  = (const float*)d_in[5];
  const float* fc2_b  = (const float*)d_in[6];
  const float* w_ih_f = (const float*)d_in[7];
  const float* w_hh_f = (const float*)d_in[8];
  const float* b_ih_f = (const float*)d_in[9];
  const float* b_hh_f = (const float*)d_in[10];
  const float* w_ih_b = (const float*)d_in[11];
  const float* w_hh_b = (const float*)d_in[12];
  const float* b_ih_b = (const float*)d_in[13];
  const float* b_hh_b = (const float*)d_in[14];
  const int* seq_idx  = (const int*)d_in[15];
  const int* lengths  = (const int*)d_in[16];
  const int n = in_sizes[0] / IN_F;

  // ---- carve workspace ----
  char* wsp = (char*)d_ws;
  size_t off = 0;
  auto carve = [&](size_t bytes) -> char* {
    char* p = wsp + off;
    off += (bytes + 255) & ~(size_t)255;
    return p;
  };
  float* h               = (float*)carve((size_t)n * HID * 4);
  float* colsum          = (float*)carve(HID * 4);
  float* colsq           = (float*)carve(HID * 4);
  float* scale           = (float*)carve(HID * 4);
  float* shift           = (float*)carve(HID * 4);
  unsigned short* seqx   = (unsigned short*)carve((size_t)SEQ * LMAX * HID * 2);
  unsigned short* seqrev = (unsigned short*)carve((size_t)SEQ * LMAX * HID * 2);
  float* xg              = (float*)carve((size_t)2 * SEQ * LMAX * G4 * 4);
  unsigned short* whhb   = (unsigned short*)carve((size_t)2 * G4 * LH * 2);
  unsigned short* hbuf0  = (unsigned short*)carve((size_t)2 * LH * LH * 2);
  unsigned short* hbuf1  = (unsigned short*)carve((size_t)2 * LH * LH * 2);
  float* cbuf            = (float*)carve((size_t)2 * LH * LH * 4);
  float* hsbuf           = (float*)carve((size_t)2 * SEQ * LMAX * LH * 4);
  unsigned short* lf     = (unsigned short*)carve((size_t)n * HID * 2);
  (void)ws_size; (void)n_in; (void)out_size;

  // ---- prep: zero state/stats, convert recurrent weights to bf16 ----
  zero_f32k<<<dim3((HID + 255) / 256), 256, 0, stream>>>(colsum, HID);
  zero_f32k<<<dim3((HID + 255) / 256), 256, 0, stream>>>(colsq, HID);
  zero_f32k<<<dim3((2 * LH * LH + 255) / 256), 256, 0, stream>>>(cbuf, 2 * LH * LH);
  zero_u16<<<dim3((2 * LH * LH + 255) / 256), 256, 0, stream>>>(hbuf0, 2 * LH * LH);
  cvt_bf16<<<dim3((G4 * LH + 255) / 256), 256, 0, stream>>>(w_hh_f, whhb, G4 * LH);
  cvt_bf16<<<dim3((G4 * LH + 255) / 256), 256, 0, stream>>>(w_hh_b, whhb + (size_t)G4 * LH, G4 * LH);

  // ---- fc1 + relu: h = relu(x @ fc1_w^T + b) ----
  gemm_wmma_bf16<false, true, false>
      <<<dim3(HID / 128, (n + 127) / 128), 256, 0, stream>>>(
          x, IN_F, fc1_w, IN_F, h, HID, n, HID, IN_F, fc1_b, nullptr);

  // ---- batch-norm stats + fused scale/shift ----
  col_stats<<<dim3((n + 127) / 128), 256, 0, stream>>>(h, n, colsum, colsq);
  bn_finalize<<<dim3(HID / 256), 256, 0, stream>>>(colsum, colsq, gamma, beta,
                                                   scale, shift, 1.f / (float)n);

  // ---- gather/normalize into bf16 sequences (fwd + clip-reversed) ----
  gather_norm<<<dim3(LMAX, SEQ), 256, 0, stream>>>(h, scale, shift, seq_idx,
                                                   lengths, seqx, seqrev);

  // ---- input-gate GEMMs: xg = seq @ w_ih^T + b_ih + b_hh ----
  gemm_wmma_bf16<true, false, true>
      <<<dim3(G4 / 128, (SEQ * LMAX) / 128), 256, 0, stream>>>(
          seqx, HID, w_ih_f, HID, xg, G4, SEQ * LMAX, G4, HID, b_ih_f, b_hh_f);
  gemm_wmma_bf16<true, false, true>
      <<<dim3(G4 / 128, (SEQ * LMAX) / 128), 256, 0, stream>>>(
          seqrev, HID, w_ih_b, HID, xg + (size_t)SEQ * LMAX * G4, G4,
          SEQ * LMAX, G4, HID, b_ih_b, b_hh_b);

  // ---- LSTM scan (both directions per launch, double-buffered h) ----
  for (int t = 0; t < LMAX; ++t) {
    const unsigned short* hin = (t & 1) ? hbuf1 : hbuf0;
    unsigned short* hout      = (t & 1) ? hbuf0 : hbuf1;
    lstm_step<<<dim3(LH / 64, SEQ / 64, 2), 256, 0, stream>>>(
        hin, hout, cbuf, xg, whhb, hsbuf, t);
  }

  // ---- scatter to bf16 lstm_feats (permutation covers every row) ----
  scatter_out<<<dim3(LMAX, SEQ), 256, 0, stream>>>(
      hsbuf, hsbuf + (size_t)SEQ * LMAX * LH, seq_idx, lengths, lf);

  // ---- fc2: logits = lstm_feats @ fc2_w^T + b -> d_out (f32) ----
  gemm_wmma_bf16<true, false, false>
      <<<dim3((NCLS + 127) / 128, (n + 127) / 128), 256, 0, stream>>>(
          lf, HID, fc2_w, HID, (float*)d_out, NCLS, n, NCLS, HID, fc2_b, nullptr);
}
